// RandFlow_Uniform_29291676959109
// MI455X (gfx1250) — matrix operand order repaired and checked by reading, not compiled, not run on hardware
//
#include <hip/hip_runtime.h>

typedef __attribute__((ext_vector_type(16))) _Float16 v16h;
typedef __attribute__((ext_vector_type(2)))  _Float16 h2;
typedef __attribute__((ext_vector_type(8)))  float    v8f;
typedef __attribute__((ext_vector_type(2)))  float    f2;

#define RAD      20
#define KW       41
#define TILE_R   64        // output rows per workgroup
#define BLK_C    64        // output cols per workgroup (4 waves x 16)
#define IN_ROWS  112       // 104 used rows, padded to 7 chunks of 16
#define IN_USED  104       // 64 + 2*20 used cols/rows of stage
#define IN_PITCH 114       // f32 pitch: 114 mod 64 = 50, gcd(50,64)=2 -> 16 distinct banks
#define IT_PITCH 118       // f16 pitch for transposed intermediate (59 dwords, gcd=1)
#define IMG_H    512
#define IMG_W    512

// normalized 1D gaussian tap; zero outside [0,40]
__device__ __forceinline__ float wgt(int idx, float invs) {
    if ((unsigned)idx > 40u) return 0.0f;
    float d = (float)(idx - RAD);
    return __expf(-d * d * 0.02f) * invs;   // 1/(2*sigma^2) = 0.02, sigma=5
}

__global__ __launch_bounds__(128)
void randflow_blur_wmma(const float* __restrict__ noise, float* __restrict__ out) {
    __shared__ float                  s_in[IN_ROWS * IN_PITCH];       // ~51 KB shared stage
    __shared__ __align__(16) _Float16 s_it[4 * 16 * IT_PITCH];        // ~15 KB, 4 wave slabs

    const int tid  = threadIdx.x;
    const int lane = tid & 31;
    const int wave = tid >> 5;

    const int cb = blockIdx.x * BLK_C - RAD;   // staged col 0 -> image col cb
    const int r0 = blockIdx.y * TILE_R;
    const int bz = blockIdx.z;
    const int b  = bz >> 1;
    const int ch = bz & 1;

    // ---- gaussian normalization (exact, once per thread) ----
    float ssum = 0.0f;
    #pragma unroll
    for (int j = 0; j < KW; ++j) {
        float d = (float)(j - RAD);
        ssum += __expf(-d * d * 0.02f);
    }
    const float invs = 1.0f / ssum;

    const float* plane = noise + (size_t)b * (IMG_H * IMG_W * 2) + ch;

    // CDNA5 speculative prefetch of the tile head
    {
        int pr = (r0 > RAD) ? (r0 - RAD) : 0;
        int pc = (cb > 0) ? cb : 0;
        __builtin_prefetch(plane + ((size_t)pr * IMG_W + pc) * 2, 0, 1);
    }

    // ---- cooperative staging: scaled flow tile (zero-padded halo) as f32 ----
    for (int idx = tid; idx < IN_ROWS * IN_PITCH; idx += 128) {
        int i  = idx / IN_PITCH;
        int j  = idx - i * IN_PITCH;
        int gr = r0 - RAD + i;
        int gc = cb + j;
        float v = 0.0f;
        if (i < IN_USED && j < IN_USED && gr >= 0 && gr < IMG_H && gc >= 0 && gc < IMG_W) {
            float x = plane[((size_t)gr * IMG_W + gc) * 2];
            v = __builtin_fmaf(2.0f, x, -1.0f);   // flow = 2*u - 1
        }
        s_in[idx] = v;
    }
    __syncthreads();

    const int nl  = lane & 15;        // N (and A-row M) index
    const int hl  = lane >> 4;        // lane half
    const int wcb = wave * 16;        // this wave's col base inside the stage
    _Float16* slab = &s_it[wave * 16 * IT_PITCH];

    // ---- H-pass banded B fragments: B[k][n] = w[k-n], K split 0..31 / 32..63 ----
    v16h bh0, bh1;
    {
        int kb = hl * 16;             // B layout: lanes 0-15 K=0..15, lanes 16-31 K=16..31
        #pragma unroll
        for (int v = 0; v < 8; ++v) {
            int k = kb + 2 * v;
            bh0[2*v]   = (_Float16)wgt(k      - nl, invs);
            bh0[2*v+1] = (_Float16)wgt(k + 1  - nl, invs);
            bh1[2*v]   = (_Float16)wgt(k + 32 - nl, invs);
            bh1[2*v+1] = (_Float16)wgt(k + 33 - nl, invs);
        }
    }

    // ---- horizontal pass: 7 chunks of 16 rows -> transposed f16 intermediate ----
    #pragma unroll
    for (int t = 0; t < 7; ++t) {
        v16h a0, a1;
        #pragma unroll
        for (int v = 0; v < 8; ++v) {
            // A layout: K = 2v + 16*half, pairs (K, K+1)
            int k = wcb + 2 * v + 16 * hl;
            const f2 p0 = *(const f2*)&s_in[(t * 16 + nl) * IN_PITCH + k];
            const f2 p1 = *(const f2*)&s_in[(t * 16 + nl) * IN_PITCH + k + 32];
            a0[2*v]   = (_Float16)p0.x;  a0[2*v+1] = (_Float16)p0.y;
            a1[2*v]   = (_Float16)p1.x;  a1[2*v+1] = (_Float16)p1.y;
        }
        v8f acc = {};
        acc = __builtin_amdgcn_wmma_f32_16x16x32_f16(false, a0, false, bh0, (short)0, acc, false, false);
        acc = __builtin_amdgcn_wmma_f32_16x16x32_f16(false, a1, false, bh1, (short)0, acc, false, false);
        #pragma unroll
        for (int j = 0; j < 8; ++j) {
            int row = t * 16 + j + 8 * hl;              // D: M = vgpr + 8*half
            slab[nl * IT_PITCH + row] = (_Float16)acc[j];    // interT[col N][row]
        }
    }
    __syncthreads();

    // ---- V-pass banded A fragments: A[m][k] = w[k-m] ----
    v16h wa0, wa1;
    #pragma unroll
    for (int v = 0; v < 8; ++v) {
        int k = 2 * v + 16 * hl;
        wa0[2*v]   = (_Float16)wgt(k      - nl, invs);
        wa0[2*v+1] = (_Float16)wgt(k + 1  - nl, invs);
        wa1[2*v]   = (_Float16)wgt(k + 32 - nl, invs);
        wa1[2*v+1] = (_Float16)wgt(k + 33 - nl, invs);
    }

    float* oplane = out + (size_t)b * (IMG_H * IMG_W * 2) + ch;
    const int c0w = blockIdx.x * BLK_C + wcb;          // this wave's output col base

    // ---- vertical pass: 4 chunks of 16 output rows ----
    #pragma unroll
    for (int r = 0; r < 4; ++r) {
        v16h b0, b1;
        int kb = hl * 16;
        #pragma unroll
        for (int v = 0; v < 8; ++v) {
            int k = r * 16 + kb + 2 * v;
            h2 q0 = *(const h2*)&slab[nl * IT_PITCH + k];
            h2 q1 = *(const h2*)&slab[nl * IT_PITCH + k + 32];
            b0[2*v] = q0.x;  b0[2*v+1] = q0.y;
            b1[2*v] = q1.x;  b1[2*v+1] = q1.y;
        }
        v8f o = {};
        o = __builtin_amdgcn_wmma_f32_16x16x32_f16(false, wa0, false, b0, (short)0, o, false, false);
        o = __builtin_amdgcn_wmma_f32_16x16x32_f16(false, wa1, false, b1, (short)0, o, false, false);
        #pragma unroll
        for (int j = 0; j < 8; ++j) {
            int orow = r0 + r * 16 + j + 8 * hl;
            oplane[((size_t)orow * IMG_W + (c0w + nl)) * 2] = o[j];
        }
    }
}

extern "C" void kernel_launch(void* const* d_in, const int* in_sizes, int n_in,
                              void* d_out, int out_size, void* d_ws, size_t ws_size,
                              hipStream_t stream) {
    (void)in_sizes; (void)n_in; (void)out_size; (void)d_ws; (void)ws_size;
    // d_in[0] = inputs (unused by the reference), d_in[1] = rand_noise [16,512,512,2] f32
    const float* noise = (const float*)d_in[1];
    float* out = (float*)d_out;

    dim3 grid(IMG_W / BLK_C, IMG_H / TILE_R, 16 * 2);  // (8, 8, 32) = 2048 workgroups
    dim3 block(128);                                    // 4 wave32 per workgroup
    hipLaunchKernelGGL(randflow_blur_wmma, grid, block, 0, stream, noise, out);
}
